// C2CFuser_498216206490
// MI455X (gfx1250) — compile-verified, matrix operand order checked by reading, not run on hardware
//
#include <hip/hip_runtime.h>
#include <hip/hip_bf16.h>
#include <math.h>

typedef __attribute__((ext_vector_type(16))) _Float16 v16h;
typedef __attribute__((ext_vector_type(8)))  _Float16 v8h;
typedef __attribute__((ext_vector_type(8)))  float    v8f;

#define LL 24
#define HH 8
#define SS 2048
#define DD 128
#define NROWS (LL*HH*SS)           /* 393216 rows per branch */
#define NT32  (NROWS/32)           /* 12288 32-row tiles per branch */
#define BWAVES 4                   /* waves per block; each wave owns 32 rows */

/* ---- workspace layout (bytes) ---- */
#define WS_WFRAG   0               /* 98304 f16 = 196608 B swizzled weights  */
#define WS_PART    196608          /* 2*12288 f32 per-tile sums = 98304 B    */
#define WS_MULT    294912          /* 2*192 f32 gate*scale     = 1536 B      */
#define WS_FUSED   296448          /* 2*NROWS*128 f32 fused tensor           */

/* ---- wfrag sub-offsets (f16 elements) ---- */
#define WF_WK 0
#define WF_WV 32768
#define WF_W1 65536
#define WF_W2 81920

/* dynamic LDS partition (f16 elements): Wp | W1 | W2 | 4x per-wave 32x128 stage */
#define LDS_WP    0
#define LDS_W1    32768
#define LDS_W2    49152
#define LDS_STG   65536
#define LDS_BYTES 163840           /* 128 KB weights + 32 KB stage */

/* =================== weight swizzle into WMMA B-fragment order ===========
 * Fragment (kc, nt) holds B[k][n] for k in [kc*32, kc*32+32), n in [nt*16, nt*16+16).
 * Lane l (col n = nt*16 + (l&15), half = l>>4) stores its 16 f16 elements
 * contiguously: e<8 -> k = kc*32 + half*8 + e ; e>=8 -> k = kc*32+16+half*8+(e-8). */
__global__ void __launch_bounds__(256) k_swizzle(
    const float* __restrict__ Wk, const float* __restrict__ Wv,
    const float* __restrict__ W1, const float* __restrict__ W2,
    _Float16* __restrict__ wfrag)
{
  int tid = blockIdx.x * 256 + threadIdx.x;       /* 0..98303 */
  const float* W; int base, local;
  if (tid < 32768)      { W = Wk; base = WF_WK; local = tid; }
  else if (tid < 65536) { W = Wv; base = WF_WV; local = tid - 32768; }
  else if (tid < 81920) { W = W1; base = WF_W1; local = tid - 65536; }
  else                  { W = W2; base = WF_W2; local = tid - 81920; }
  int e    = local & 15;
  int lane = (local >> 4) & 31;
  int f    = local >> 9;                           /* fragment = kc*8 + nt */
  int nt   = f & 7, kc = f >> 3;
  int half = lane >> 4;
  int k = kc*32 + ((e < 8) ? (half*8 + e) : (16 + half*8 + (e - 8)));
  int n = nt*16 + (lane & 15);
  wfrag[base + local] = (_Float16)W[k*128 + n];
}

/* =================== helpers =================== */
__device__ inline v16h cat8(v8h lo, v8h hi) {
  v16h r;
  #pragma unroll
  for (int i = 0; i < 8; ++i) { r[i] = lo[i]; r[8+i] = hi[i]; }
  return r;
}

/* load + convert one 16x32 A fragment from a global f32 row-major matrix */
__device__ inline v16h load_a_global(const float* __restrict__ p) {
  float4 q0 = *(const float4*)(p);
  float4 q1 = *(const float4*)(p + 4);
  float4 q2 = *(const float4*)(p + 16);
  float4 q3 = *(const float4*)(p + 20);
  v16h a;
  a[0]=(_Float16)q0.x;  a[1]=(_Float16)q0.y;  a[2]=(_Float16)q0.z;  a[3]=(_Float16)q0.w;
  a[4]=(_Float16)q1.x;  a[5]=(_Float16)q1.y;  a[6]=(_Float16)q1.z;  a[7]=(_Float16)q1.w;
  a[8]=(_Float16)q2.x;  a[9]=(_Float16)q2.y;  a[10]=(_Float16)q2.z; a[11]=(_Float16)q2.w;
  a[12]=(_Float16)q3.x; a[13]=(_Float16)q3.y; a[14]=(_Float16)q3.z; a[15]=(_Float16)q3.w;
  return a;
}

/* load one 16x32 A fragment from the LDS stage (row base srow, f16 row pitch 128) */
__device__ inline v16h load_a_lds(const _Float16* __restrict__ srow, int kc, int half) {
  v8h a0 = *(const v8h*)(srow + kc*32 + half*8);
  v8h a1 = *(const v8h*)(srow + kc*32 + 16 + half*8);
  return cat8(a0, a1);
}

/* =================== pass 1: fused = MLP(concat(x_r,x_s)) =================
 * One wave owns a 32-row tile (two 16-row A blocks); each LDS B fragment
 * feeds two back-to-back WMMAs.                                            */
__global__ void __launch_bounds__(128) k_fuse(
    const float* __restrict__ KR, const float* __restrict__ KS,
    const float* __restrict__ VR, const float* __restrict__ VS,
    const float* __restrict__ bk, const float* __restrict__ bv,
    const float* __restrict__ b1g, const float* __restrict__ b2g,
    const _Float16* __restrict__ wfrag,
    float* __restrict__ fused, float* __restrict__ part)
{
  extern __shared__ char smem[];
  _Float16* wp_l = (_Float16*)smem + LDS_WP;
  _Float16* w1_l = (_Float16*)smem + LDS_W1;
  _Float16* w2_l = (_Float16*)smem + LDS_W2;
  _Float16* stg  = (_Float16*)smem + LDS_STG;

  const int wave   = threadIdx.x >> 5;             /* 0..3 */
  const int lane   = threadIdx.x & 31;
  const int half   = lane >> 4;
  const int lr     = lane & 15;
  const int branch = blockIdx.y;
  const int tile   = blockIdx.x * BWAVES + wave;   /* 32-row tile id */
  const int row0   = tile << 5;

  /* ---- cooperative weight staging: branch Wp (64KB) + W1|W2 (64KB) ---- */
  {
    const float4* s0 = (const float4*)(wfrag + (branch ? WF_WV : WF_WK));
    const float4* s1 = (const float4*)(wfrag + WF_W1);   /* W1,W2 contiguous */
    float4* d0 = (float4*)wp_l;
    float4* d1 = (float4*)w1_l;
    for (int i = threadIdx.x; i < 4096; i += 128) { d0[i] = s0[i]; d1[i] = s1[i]; }
  }
  __syncthreads();

  const float* xr = branch ? VR : KR;
  const float* xs = branch ? VS : KS;
  const float* bp = branch ? bv : bk;
  const int rowA = row0 + lr;                      /* rows of A block 0 */
  const int rowB = rowA + 16;                      /* rows of A block 1 */

  /* speculative prefetch of the next block's input rows */
  __builtin_prefetch(xr + (size_t)(rowA + 128)*128, 0, 1);
  __builtin_prefetch(xs + (size_t)(rowA + 128)*128, 0, 1);

  v8f zero;
  #pragma unroll
  for (int i = 0; i < 8; ++i) zero[i] = 0.0f;

  /* ---- GEMM1: (32x256) x (256x128) ---- */
  v8f acc[16];
  #pragma unroll
  for (int i = 0; i < 16; ++i) acc[i] = zero;
  #pragma unroll
  for (int kc = 0; kc < 8; ++kc) {
    const float* src = (kc < 4) ? xr : xs;
    const int c0 = (kc & 3)*32 + half*8;
    v16h a0 = load_a_global(src + (size_t)rowA * 128 + c0);
    v16h a1 = load_a_global(src + (size_t)rowB * 128 + c0);
    #pragma unroll
    for (int nt = 0; nt < 8; ++nt) {
      v16h b = *(const v16h*)(wp_l + (size_t)(kc*8 + nt)*512 + lane*16);
      acc[nt]   = __builtin_amdgcn_wmma_f32_16x16x32_f16(
          false, a0, false, b, (short)0, acc[nt],   false, false);
      acc[8+nt] = __builtin_amdgcn_wmma_f32_16x16x32_f16(
          false, a1, false, b, (short)0, acc[8+nt], false, false);
    }
  }

  /* bias + stage x (D-layout -> row-major f16) into this wave's 32x128 tile */
  _Float16* srow_w = stg + wave*4096;
  #pragma unroll
  for (int nt = 0; nt < 8; ++nt) {
    float b0 = bp[nt*16 + lr];
    #pragma unroll
    for (int r = 0; r < 8; ++r) {
      srow_w[(r + 8*half)*128      + nt*16 + lr] = (_Float16)(acc[nt][r]   + b0);
      srow_w[(16 + r + 8*half)*128 + nt*16 + lr] = (_Float16)(acc[8+nt][r] + b0);
    }
  }

  /* ---- GEMM2: x @ W1 + b1, then exact GELU ---- */
  v8f acc2[16];
  #pragma unroll
  for (int i = 0; i < 16; ++i) acc2[i] = zero;
  const _Float16* srow_r0 = stg + wave*4096 + lr*128;
  const _Float16* srow_r1 = srow_r0 + 16*128;
  #pragma unroll
  for (int kc = 0; kc < 4; ++kc) {
    v16h a0 = load_a_lds(srow_r0, kc, half);
    v16h a1 = load_a_lds(srow_r1, kc, half);
    #pragma unroll
    for (int nt = 0; nt < 8; ++nt) {
      v16h b = *(const v16h*)(w1_l + (size_t)(kc*8 + nt)*512 + lane*16);
      acc2[nt]   = __builtin_amdgcn_wmma_f32_16x16x32_f16(
          false, a0, false, b, (short)0, acc2[nt],   false, false);
      acc2[8+nt] = __builtin_amdgcn_wmma_f32_16x16x32_f16(
          false, a1, false, b, (short)0, acc2[8+nt], false, false);
    }
  }
  #pragma unroll
  for (int nt = 0; nt < 8; ++nt) {
    float b0 = b1g[nt*16 + lr];
    #pragma unroll
    for (int r = 0; r < 8; ++r) {
      float v0 = acc2[nt][r]   + b0;
      float v1 = acc2[8+nt][r] + b0;
      float g0 = 0.5f * v0 * (1.0f + erff(v0 * 0.70710678118f));  /* exact GELU */
      float g1 = 0.5f * v1 * (1.0f + erff(v1 * 0.70710678118f));
      srow_w[(r + 8*half)*128      + nt*16 + lr] = (_Float16)g0;
      srow_w[(16 + r + 8*half)*128 + nt*16 + lr] = (_Float16)g1;
    }
  }

  /* ---- GEMM3: h @ W2 + b2 ---- */
  v8f acc3[16];
  #pragma unroll
  for (int i = 0; i < 16; ++i) acc3[i] = zero;
  #pragma unroll
  for (int kc = 0; kc < 4; ++kc) {
    v16h a0 = load_a_lds(srow_r0, kc, half);
    v16h a1 = load_a_lds(srow_r1, kc, half);
    #pragma unroll
    for (int nt = 0; nt < 8; ++nt) {
      v16h b = *(const v16h*)(w2_l + (size_t)(kc*8 + nt)*512 + lane*16);
      acc3[nt]   = __builtin_amdgcn_wmma_f32_16x16x32_f16(
          false, a0, false, b, (short)0, acc3[nt],   false, false);
      acc3[8+nt] = __builtin_amdgcn_wmma_f32_16x16x32_f16(
          false, a1, false, b, (short)0, acc3[8+nt], false, false);
    }
  }

  /* ---- bias + write fused tile + deterministic per-tile pooled sum ---- */
  float lsum = 0.0f;
  const size_t obase = (size_t)branch * NROWS * 128;
  #pragma unroll
  for (int nt = 0; nt < 8; ++nt) {
    float b0 = b2g[nt*16 + lr];
    #pragma unroll
    for (int r = 0; r < 8; ++r) {
      float v0 = acc3[nt][r]   + b0;
      float v1 = acc3[8+nt][r] + b0;
      lsum += v0 + v1;
      int m = r + 8*half;
      fused[obase + (size_t)(row0 + m)*128      + nt*16 + lr] = v0;
      fused[obase + (size_t)(row0 + 16 + m)*128 + nt*16 + lr] = v1;
    }
  }
  #pragma unroll
  for (int off = 16; off > 0; off >>= 1) lsum += __shfl_xor(lsum, off, 32);
  if (lane == 0) part[branch * NT32 + tile] = lsum;
}

/* =================== threefry2x32 (JAX key 42, n=24) ====================== */
__device__ inline unsigned rotl32(unsigned x, int d) { return (x << d) | (x >> (32 - d)); }

__device__ inline float tf_uniform24(int i) {
  const unsigned k0 = 0u, k1 = 42u;
  const unsigned k2 = k0 ^ k1 ^ 0x1BD11BDAu;
  bool hi = (i >= 12);
  unsigned x0 = hi ? (unsigned)(i - 12) : (unsigned)i;
  unsigned x1 = hi ? (unsigned)i        : (unsigned)(i + 12);
  x0 += k0; x1 += k1;
  const int RA[4] = {13, 15, 26, 6};
  const int RB[4] = {17, 29, 16, 24};
  #pragma unroll
  for (int j = 0; j < 4; ++j) { x0 += x1; x1 = rotl32(x1, RA[j]); x1 ^= x0; }
  x0 += k1; x1 += k2 + 1u;
  #pragma unroll
  for (int j = 0; j < 4; ++j) { x0 += x1; x1 = rotl32(x1, RB[j]); x1 ^= x0; }
  x0 += k2; x1 += k0 + 2u;
  #pragma unroll
  for (int j = 0; j < 4; ++j) { x0 += x1; x1 = rotl32(x1, RA[j]); x1 ^= x0; }
  x0 += k0; x1 += k1 + 3u;
  #pragma unroll
  for (int j = 0; j < 4; ++j) { x0 += x1; x1 = rotl32(x1, RB[j]); x1 ^= x0; }
  x0 += k1; x1 += k2 + 4u;
  #pragma unroll
  for (int j = 0; j < 4; ++j) { x0 += x1; x1 = rotl32(x1, RA[j]); x1 ^= x0; }
  x0 += k2; x1 += k0 + 5u;
  unsigned bits = hi ? x1 : x0;
  return __uint_as_float((bits >> 9) | 0x3f800000u) - 1.0f;
}

/* =================== pass 2: pooled -> sigmoid scale * gumbel gate ======== */
__global__ void k_gate(const float* __restrict__ part,
                       const float* __restrict__ Wdw, const float* __restrict__ bdw,
                       const float* __restrict__ gate_param,
                       float* __restrict__ mult)
{
  __shared__ float pool[384];
  int t = threadIdx.x;                       /* 0..383 = [branch][l][h] */
  {
    int b = t / 192, lh = t % 192;
    float s = 0.0f;
    for (int i = 0; i < 64; ++i) s += part[b*NT32 + lh*64 + i];
    pool[t] = s * (1.0f / ((float)SS * 128.0f));
  }
  __syncthreads();
  int b = t / 192, rem = t % 192, l = rem / 8, g = rem % 8;
  float acc = bdw[g];
  #pragma unroll
  for (int h = 0; h < 8; ++h) acc += pool[b*192 + l*8 + h] * Wdw[g*8 + h];
  float scale = 1.0f / (1.0f + expf(-acc));
  float u   = tf_uniform24(l);
  float gum = -logf(-logf(u + 1e-10f) + 1e-10f);
  float gate = 1.0f / (1.0f + expf(-(gate_param[l] + gum) * 2.0f));  /* /T, T=0.5 */
  mult[t] = gate * scale;
}

/* =================== pass 3: out = x_r + mult * fused ===================== */
__global__ void __launch_bounds__(256) k_out(
    const float* __restrict__ KR, const float* __restrict__ VR,
    const float* __restrict__ fused, const float* __restrict__ mult,
    float* __restrict__ out)
{
  size_t i4  = (size_t)blockIdx.x * 256 + threadIdx.x;
  size_t idx = i4 * 4;
  const size_t C = (size_t)NROWS * 128;
  int branch = (idx >= C) ? 1 : 0;
  size_t rem = idx - (branch ? C : 0);
  int lh = (int)(rem >> 18);                 /* rem / (S*128) = (l*H+h)      */
  float m = mult[branch*192 + lh];
  const float* xr = (branch ? VR : KR) + rem;
  float4 a = *(const float4*)(xr);
  float4 b = *(const float4*)(fused + idx);
  float4 o;
  o.x = a.x + m*b.x; o.y = a.y + m*b.y; o.z = a.z + m*b.z; o.w = a.w + m*b.w;
  *(float4*)(out + idx) = o;
}

/* ========================================================================= */
extern "C" void kernel_launch(void* const* d_in, const int* in_sizes, int n_in,
                              void* d_out, int out_size, void* d_ws, size_t ws_size,
                              hipStream_t stream)
{
  const float* K_R = (const float*)d_in[0];
  const float* V_R = (const float*)d_in[1];
  const float* K_S = (const float*)d_in[2];
  const float* V_S = (const float*)d_in[3];
  const float* Wk  = (const float*)d_in[4];
  const float* bk  = (const float*)d_in[5];
  const float* Wv  = (const float*)d_in[6];
  const float* bv  = (const float*)d_in[7];
  const float* W1  = (const float*)d_in[8];
  const float* b1  = (const float*)d_in[9];
  const float* W2  = (const float*)d_in[10];
  const float* b2  = (const float*)d_in[11];
  const float* Wdw = (const float*)d_in[12];
  const float* bdw = (const float*)d_in[13];
  const float* gp  = (const float*)d_in[14];

  char* ws = (char*)d_ws;
  _Float16* wfrag = (_Float16*)(ws + WS_WFRAG);
  float* part  = (float*)(ws + WS_PART);
  float* mult  = (float*)(ws + WS_MULT);
  float* fused = (float*)(ws + WS_FUSED);
  float* out   = (float*)d_out;

  k_swizzle<<<384, 256, 0, stream>>>(Wk, Wv, W1, W2, wfrag);
  k_fuse<<<dim3(NROWS/128, 2), 128, LDS_BYTES, stream>>>(
      K_R, K_S, V_R, V_S, bk, bv, b1, b2, wfrag, fused, part);
  k_gate<<<1, 384, 0, stream>>>(part, Wdw, bdw, gp, mult);
  k_out<<<98304, 256, 0, stream>>>(K_R, V_R, fused, mult, out);
}
